// NodeDenoiser_27951647162967
// MI455X (gfx1250) — matrix operand order repaired and checked by reading, not compiled, not run on hardware
//
#include <hip/hip_runtime.h>
#include <cstdint>
#include <cstddef>

// Problem constants (match reference)
#define ZB 4
#define NBN 2048
#define KNB 32
#define DB 128
#define HHD 4
#define ROWS (ZB * NBN)   // 8192 token rows

typedef __attribute__((ext_vector_type(16))) __bf16 v16bf;
typedef __attribute__((ext_vector_type(8)))  float  v8f;

// TDM descriptor vector types (per CDNA5_HIP.md probe signatures)
typedef __attribute__((ext_vector_type(4))) unsigned int tdm_u32x4;
typedef __attribute__((ext_vector_type(8))) int          tdm_i32x8;
typedef __attribute__((ext_vector_type(4))) int          tdm_i32x4;

#if defined(__has_builtin)
#if __has_builtin(__builtin_amdgcn_tensor_load_to_lds) && \
    __has_builtin(__builtin_amdgcn_s_wait_tensorcnt)
#define HAVE_TDM 1
#endif
#endif
#ifndef HAVE_TDM
#define HAVE_TDM 0
#endif

// ---------------- scalar helpers ----------------
__device__ __forceinline__ __bf16 f2bf(float x) {
  union { float f; unsigned u; } v; v.f = x;
  unsigned r = v.u + 0x7FFFu + ((v.u >> 16) & 1u);   // round-to-nearest-even
  union { unsigned short s; __bf16 b; } o; o.s = (unsigned short)(r >> 16);
  return o.b;
}
__device__ __forceinline__ float bf2f(__bf16 x) {
  union { unsigned short s; __bf16 b; } i; i.b = x;
  union { unsigned u; float f; } o; o.u = ((unsigned)i.s) << 16;
  return o.f;
}
__device__ __forceinline__ float dsilu(float x) { return x / (1.f + __expf(-x)); }

// ---------------- WMMA helpers ----------------
__device__ __forceinline__ v8f wmma_bf16(v16bf a, v16bf b, v8f c) {
  // v_wmma_f32_16x16x32_bf16  D = A(16x32) * B(32x16) + C(16x16 f32)
  return __builtin_amdgcn_wmma_f32_16x16x32_bf16(false, a, false, b, (short)0, c,
                                                 false, false);
}

// A fragment (16x32 bf16) from a row-major bf16 tile with leading dim ldk.
// ISA layout: lanes 0-15 row m, K pairs {0..7 | 16..23}; lanes 16-31 K+8.
// Each lane's 16 elements are two contiguous 16B runs -> b128 loads.
__device__ __forceinline__ v16bf load_afrag_bf(const __bf16* base, int ldk, int kt,
                                               int lane) {
  int m = lane & 15, hi = lane >> 4;
  const __bf16* r = base + (size_t)m * ldk + kt * 32 + (hi ? 8 : 0);
  v16bf a;
#pragma unroll
  for (int j = 0; j < 8; ++j) {
    int kb = ((j < 4) ? 0 : 16) + 2 * (j & 3);
    a[2 * j]     = r[kb];
    a[2 * j + 1] = r[kb + 1];
  }
  return a;
}

// A fragment from a row-major *f32* tile (LDS-staged edges), converted to bf16.
__device__ __forceinline__ v16bf load_afrag_f32(const float* base, int ldk, int kt,
                                                int lane) {
  int m = lane & 15, hi = lane >> 4;
  const float* r = base + (size_t)m * ldk + kt * 32 + (hi ? 8 : 0);
  v16bf a;
#pragma unroll
  for (int j = 0; j < 8; ++j) {
    int kb = ((j < 4) ? 0 : 16) + 2 * (j & 3);
    a[2 * j]     = f2bf(r[kb]);
    a[2 * j + 1] = f2bf(r[kb + 1]);
  }
  return a;
}

// B fragment (32x16 bf16) from pre-swizzled weights: 32 bytes contiguous per lane.
__device__ __forceinline__ v16bf load_bfrag(const __bf16* Bfr, int f, int lane) {
  return *(const v16bf*)(Bfr + ((size_t)f * 32 + lane) * 16);
}

// ---------------- Tensor Data Mover: 1-D f32 tile -> LDS ----------------
#if HAVE_TDM
__device__ __forceinline__ void tdm_load_tile_f32(unsigned ldsByteAddr,
                                                  const float* gsrc,
                                                  unsigned elems) {
  unsigned long long ga = (unsigned long long)(size_t)gsrc;
  unsigned galo = __builtin_amdgcn_readfirstlane((unsigned)ga);
  unsigned gahi = __builtin_amdgcn_readfirstlane((unsigned)(ga >> 32));
  unsigned lb   = __builtin_amdgcn_readfirstlane(ldsByteAddr);
  tdm_u32x4 g0;
  g0[0] = 1u;                                   // count=1, user descriptor
  g0[1] = lb;                                   // lds_addr (bytes)
  g0[2] = galo;                                 // global_addr[31:0]
  g0[3] = (gahi & 0x1FFFFFFu) | (2u << 30);     // global_addr[56:32] | type=2
  tdm_i32x8 g1;
  g1[0] = (int)(2u << 16);                      // data_size=4B, no multicast
  g1[1] = (int)((elems & 0xFFFFu) << 16);       // tensor_dim0[15:0] @ bits63:48
  g1[2] = (int)((elems >> 16) & 0xFFFFu);       // tensor_dim0[31:16]; tensor_dim1=0
  g1[3] = (int)((elems & 0xFFFFu) << 16);       // tile_dim0 @ bits127:112
  g1[4] = 0;                                    // tile_dim1/2 unused (1-D)
  g1[5] = (int)elems;                           // tensor_dim0_stride lo
  g1[6] = 0;
  g1[7] = 0;
  tdm_i32x4 gz = {0, 0, 0, 0};
#if defined(__clang_major__) && __clang_major__ >= 23
  tdm_i32x8 gx = {0, 0, 0, 0, 0, 0, 0, 0};
  __builtin_amdgcn_tensor_load_to_lds(g0, g1, gz, gz, gx, 0);
#else
  __builtin_amdgcn_tensor_load_to_lds(g0, g1, gz, gz, 0);
#endif
}
#endif

// ---------------- weight pre-swizzle ----------------
// Converts f32 weights to bf16 fragments. headMode=1: W is [H=4][128][32] and the
// logical matrix is the head-concat [128][128].
__global__ __launch_bounds__(256) void swizzle_b_kernel(const float* __restrict__ W,
                                                        __bf16* __restrict__ out,
                                                        int kdim, int ncols,
                                                        int headMode) {
  int tid = blockIdx.x * 256 + threadIdx.x;
  int ntiles = ncols >> 4;
  int total = (kdim >> 5) * ntiles * 256;
  if (tid >= total) return;
  int j    = tid & 7;
  int lane = (tid >> 3) & 31;
  int f    = tid >> 8;
  int kt = f / ntiles, nt = f % ntiles;
  int n  = nt * 16 + (lane & 15);
  int k  = kt * 32 + (lane >> 4) * 8 + ((j < 4) ? 0 : 16) + 2 * (j & 3);
  float w0, w1;
  if (headMode) {
    int h = n >> 5, e = n & 31;
    w0 = W[((size_t)h * kdim + k) * 32 + e];
    w1 = W[((size_t)h * kdim + k + 1) * 32 + e];
  } else {
    w0 = W[(size_t)k * ncols + n];
    w1 = W[(size_t)(k + 1) * ncols + n];
  }
  __bf16* dst = out + ((size_t)f * 32 + lane) * 16 + 2 * j;
  dst[0] = f2bf(w0);
  dst[1] = f2bf(w1);
}

__global__ __launch_bounds__(256) void cvt_f32_bf16_kernel(const float* __restrict__ in,
                                                           __bf16* __restrict__ out,
                                                           int n) {
  int i = blockIdx.x * 256 + threadIdx.x;
  if (i < n) out[i] = f2bf(in[i]);
}

// ---------------- generic WMMA GEMM (software-pipelined k loop) ----------------
// out = epilogue( A[rows x kdim](bf16 row-major) @ Bfrag + bias )
// epilogue: optional SiLU; optional residual  val = resAdd + resScale * val.
__global__ __launch_bounds__(256) void gemm_tile_kernel(
    const __bf16* __restrict__ A, const __bf16* __restrict__ Bfr,
    const float* __restrict__ bias, int rows, int kdim, int ncols, int act_silu,
    float* __restrict__ outF, __bf16* __restrict__ outB,
    const float* __restrict__ resAdd, const float* __restrict__ resScale,
    int rsStride, int rsOff) {
  int lane = threadIdx.x & 31;
  int wave = threadIdx.x >> 5;
  int ntiles = ncols >> 4;
  int tiles  = (rows >> 4) * ntiles;
  int tile = blockIdx.x * 8 + wave;
  if (tile >= tiles) return;                 // wave-uniform; EXEC stays all-ones
  int rt = tile / ntiles, nt = tile % ntiles;
  int ktiles = kdim >> 5;
  v8f c = {0.f, 0.f, 0.f, 0.f, 0.f, 0.f, 0.f, 0.f};
  const __bf16* Ab = A + (size_t)rt * 16 * kdim;
  v16bf a = load_afrag_bf(Ab, kdim, 0, lane);
  v16bf b = load_bfrag(Bfr, nt, lane);
  for (int kt = 0; kt < ktiles; ++kt) {
    v16bf an = a, bn = b;
    if (kt + 1 < ktiles) {                   // prefetch next fragments
      an = load_afrag_bf(Ab, kdim, kt + 1, lane);
      bn = load_bfrag(Bfr, (kt + 1) * ntiles + nt, lane);
    }
    c = wmma_bf16(a, b, c);
    a = an; b = bn;
  }
  int n = nt * 16 + (lane & 15);
  int hi = lane >> 4;
  float bv = bias ? bias[n] : 0.f;
#pragma unroll
  for (int v = 0; v < 8; ++v) {
    int m = v + (hi ? 8 : 0);
    size_t r = (size_t)rt * 16 + m;
    float val = c[v] + bv;
    if (act_silu) val = dsilu(val);
    if (resAdd) val = resAdd[r * ncols + n] + resScale[r * rsStride + rsOff + n] * val;
    if (outF) outF[r * ncols + n] = val;
    if (outB) outB[r * ncols + n] = f2bf(val);
  }
}

// ---------------- StaticLayerNorm + FiLM modulation ----------------
// out(bf16) = gamma * ((x - mean)/std_ddof1) + beta ; std==0 -> 1
__global__ __launch_bounds__(256) void sln_mod_kernel(
    const float* __restrict__ x, const float* __restrict__ gamma, int gStride,
    int gOff, const float* __restrict__ beta, __bf16* __restrict__ out, int rows) {
  int lane = threadIdx.x & 31;
  int wave = threadIdx.x >> 5;
  int row = blockIdx.x * 8 + wave;
  if (row >= rows) return;
  float v[4];
  float s = 0.f;
#pragma unroll
  for (int j = 0; j < 4; ++j) {
    v[j] = x[(size_t)row * DB + lane * 4 + j];
    s += v[j];
  }
  for (int m = 16; m; m >>= 1) s += __shfl_xor(s, m, 32);
  float mean = s * (1.f / 128.f);
  float ss = 0.f;
#pragma unroll
  for (int j = 0; j < 4; ++j) {
    v[j] -= mean;
    ss += v[j] * v[j];
  }
  for (int m = 16; m; m >>= 1) ss += __shfl_xor(ss, m, 32);
  float stdv = sqrtf(ss * (1.f / 127.f));
  float inv = (stdv == 0.f) ? 1.f : (1.f / stdv);
#pragma unroll
  for (int j = 0; j < 4; ++j) {
    int col = lane * 4 + j;
    float g = gamma[(size_t)row * gStride + gOff + col];
    float b = beta[(size_t)row * DB + col];
    out[(size_t)row * DB + col] = f2bf(g * v[j] * inv + b);
  }
}

// ---------------- FiLM MLP layer on a 16-token tile (all WMMA) ----------------
// kdim fixed to 128: 4 A fragments hoisted and reused across all n-tiles.
__device__ __forceinline__ void film_layer_bf(const __bf16* Ain, const __bf16* Wfr,
                                              const float* bias, __bf16* outLds,
                                              int ncols, int lane, int dosilu) {
  int ntiles = ncols >> 4;
  int n16 = lane & 15, hi = lane >> 4;
  v16bf a4[4];
#pragma unroll
  for (int kt = 0; kt < 4; ++kt) a4[kt] = load_afrag_bf(Ain, 128, kt, lane);
  for (int nt = 0; nt < ntiles; ++nt) {
    v8f c = {0.f, 0.f, 0.f, 0.f, 0.f, 0.f, 0.f, 0.f};
#pragma unroll
    for (int kt = 0; kt < 4; ++kt)
      c = wmma_bf16(a4[kt], load_bfrag(Wfr, kt * ntiles + nt, lane), c);
    int n = nt * 16 + n16;
    float bv = bias[n];
#pragma unroll
    for (int v = 0; v < 8; ++v) {
      int m = v + (hi ? 8 : 0);
      float val = c[v] + bv;
      if (dosilu) val = dsilu(val);
      outLds[m * ncols + n] = f2bf(val);
    }
  }
}

// First FiLM layer reading the TDM-staged f32 edge tile from LDS.
__device__ __forceinline__ void film_layer_l1(const float* Etile,
                                              const __bf16* Wfr,
                                              const float* bias,
                                              __bf16* outLds, int lane) {
  int n16 = lane & 15, hi = lane >> 4;
  v16bf a4[4];
#pragma unroll
  for (int kt = 0; kt < 4; ++kt) a4[kt] = load_afrag_f32(Etile, DB, kt, lane);
  for (int nt = 0; nt < 8; ++nt) {
    v8f c = {0.f, 0.f, 0.f, 0.f, 0.f, 0.f, 0.f, 0.f};
#pragma unroll
    for (int kt = 0; kt < 4; ++kt)
      c = wmma_bf16(a4[kt], load_bfrag(Wfr, kt * 8 + nt, lane), c);
    int n = nt * 16 + n16;
    float bv = bias[n];
#pragma unroll
    for (int v = 0; v < 8; ++v) {
      int m = v + (hi ? 8 : 0);
      outLds[m * DB + n] = f2bf(dsilu(c[v] + bv));
    }
  }
}

// ---------------- fused FiLM + attention kernel ----------------
// One wave per node; 4 waves / workgroup. Per-wave LDS: compute scratch (46080 B)
// + double-buffered TDM edge staging (2 x 16 KB).
#define AT_COMPUTE_BYTES 46080
#define AT_WAVE_BYTES (AT_COMPUTE_BYTES + 2 * KNB * DB * 4)   // 78848
#define AT_WAVES 4
#define AT_WGS 64
#define NODES_PER_WAVE 32

__global__ __launch_bounds__(128) void film_attn_kernel(
    const float* __restrict__ edges, const int* __restrict__ nbrs,
    const unsigned char* __restrict__ nmask, const __bf16* __restrict__ nodesI,
    const float* __restrict__ Qbuf,
    const __bf16* WK1, const float* bK1, const __bf16* WK2, const float* bK2,
    const __bf16* WK3, const float* bK3,
    const __bf16* WV1, const float* bV1, const __bf16* WV2, const float* bV2,
    const __bf16* WV3, const float* bV3,
    const __bf16* Wkc, const float* bkc, const __bf16* Wvc, const float* bvc,
    __bf16* __restrict__ attnOut) {
  extern __shared__ char smem[];
  int lane = threadIdx.x & 31;
  int wave = threadIdx.x >> 5;
  char* base = smem + (size_t)wave * AT_WAVE_BYTES;
  __bf16* njc  = (__bf16*)base;          // 32x128 gathered neighbor features
  __bf16* hb   = njc  + 32 * DB;         // 16x128 hidden1
  __bf16* h2   = hb   + 16 * DB;         // 16x128 hidden2
  __bf16* gbuf = h2   + 16 * DB;         // 16x256 g|b
  __bf16* kf   = gbuf + 16 * 256;        // 16x128 film output
  __bf16* Kt   = kf   + 16 * DB;         // 32x128 keys
  __bf16* Vt   = Kt   + 32 * DB;         // 32x128 values
  float*  qs   = (float*)(Vt + 32 * DB); // 128 query
  float*  ps   = qs + 128;               // 4x32 softmax probs
  float*  est0 = ps + 128;               // 32x128 f32 edge staging (buf 0)
  float*  est1 = est0 + KNB * DB;        // 32x128 f32 edge staging (buf 1)
  float*  ebufs[2] = {est0, est1};

  int waveGlobal = blockIdx.x * AT_WAVES + wave;
  int node0 = waveGlobal * NODES_PER_WAVE;
#if HAVE_TDM
  // Kick off the first async edge-tile DMA (16 KB contiguous) into LDS.
  tdm_load_tile_f32((unsigned)(size_t)est0,
                    edges + (size_t)node0 * KNB * DB, KNB * DB);
#endif
  for (int it = 0; it < NODES_PER_WAVE; ++it) {
    int node = node0 + it;                         // [0, Z*N)
    int zbase = (node >> 11) << 11;                // z * N
    float* ebuf = ebufs[it & 1];
#if HAVE_TDM
    if (it + 1 < NODES_PER_WAVE) {
      // overlap next node's DMA with this node's 640 WMMAs
      tdm_load_tile_f32((unsigned)(size_t)ebufs[(it + 1) & 1],
                        edges + (size_t)(node + 1) * KNB * DB, KNB * DB);
      __builtin_amdgcn_s_wait_tensorcnt(1);        // current buffer is ready
    } else {
      __builtin_amdgcn_s_wait_tensorcnt(0);
    }
#else
    for (int i = lane; i < KNB * DB / 4; i += 32)
      ((uint4*)ebuf)[i] = ((const uint4*)(edges + (size_t)node * KNB * DB))[i];
#endif

    // gather neighbor rows of modulated nodes (bf16), lane k -> neighbor k
    int nIdx = zbase + nbrs[(size_t)node * KNB + lane];
    {
      const uint4* s = (const uint4*)(nodesI + (size_t)nIdx * DB);
      uint4* d = (uint4*)(njc + lane * DB);
#pragma unroll
      for (int j = 0; j < 16; ++j) d[j] = s[j];
    }
    __syncthreads();

    // FiLM MLP + K/V projection, branches br=0:K, 1:V; two 16-edge tiles each
    for (int br = 0; br < 2; ++br) {
      const __bf16* W1 = br ? WV1 : WK1;
      const __bf16* W2 = br ? WV2 : WK2;
      const __bf16* W3 = br ? WV3 : WK3;
      const float* b1 = br ? bV1 : bK1;
      const float* b2 = br ? bV2 : bK2;
      const float* b3 = br ? bV3 : bK3;
      const __bf16* Wc = br ? Wvc : Wkc;
      const float* bc = br ? bvc : bkc;
      __bf16* KV = br ? Vt : Kt;
      for (int tt = 0; tt < 2; ++tt) {
        film_layer_l1(ebuf + (size_t)tt * 16 * DB, W1, b1, hb, lane);
        __syncthreads();
        film_layer_bf(hb, W2, b2, h2, DB, lane, 1);
        __syncthreads();
        film_layer_bf(h2, W3, b3, gbuf, 256, lane, 0);
        __syncthreads();
        // kf = g * nodes_j + b
        for (int idx = lane; idx < 16 * DB; idx += 32) {
          int m = idx >> 7, d = idx & 127;
          float g = bf2f(gbuf[m * 256 + d]);
          float bb = bf2f(gbuf[m * 256 + 128 + d]);
          float xv = bf2f(njc[(tt * 16 + m) * DB + d]);
          kf[m * DB + d] = f2bf(g * xv + bb);
        }
        __syncthreads();
        film_layer_bf(kf, Wc, bc, KV + tt * 16 * DB, DB, lane, 0);
        __syncthreads();
      }
    }

    // attention: lane k owns neighbor k; heads h = 0..3, Dk = 32
#pragma unroll
    for (int j = 0; j < 4; ++j)
      qs[lane * 4 + j] = Qbuf[(size_t)node * DB + lane * 4 + j];
    __syncthreads();

    float sc[HHD];
#pragma unroll
    for (int h = 0; h < HHD; ++h) {
      float acc = 0.f;
      for (int e = 0; e < 32; ++e)
        acc += qs[h * 32 + e] * bf2f(Kt[lane * DB + h * 32 + e]);
      sc[h] = acc * 0.17677669529663687f;   // 1/sqrt(Dk)
    }
    int inv = nmask[(size_t)node * KNB + lane] ? 0 : 1;
    int cnt = inv;
    for (int m = 16; m; m >>= 1) cnt += __shfl_xor(cnt, m, 32);
    if (inv && cnt < 32) {
#pragma unroll
      for (int h = 0; h < HHD; ++h) sc[h] = -INFINITY;
    }
#pragma unroll
    for (int h = 0; h < HHD; ++h) {
      float mx = sc[h];
      for (int m = 16; m; m >>= 1) mx = fmaxf(mx, __shfl_xor(mx, m, 32));
      float e = __expf(sc[h] - mx);
      float sum = e;
      for (int m = 16; m; m >>= 1) sum += __shfl_xor(sum, m, 32);
      ps[h * 32 + lane] = e / sum;
    }
    __syncthreads();
    // out[h][e] = sum_k P[h][k] * V[k][h*32+e]; store permuted (e*H + h)
#pragma unroll
    for (int j = 0; j < 4; ++j) {
      int d = lane * 4 + j, h = d >> 5, e = d & 31;
      float acc = 0.f;
      for (int k = 0; k < 32; ++k) acc += ps[h * 32 + k] * bf2f(Vt[k * DB + d]);
      attnOut[(size_t)node * DB + e * HHD + h] = f2bf(acc);
    }
    __syncthreads();
  }
}

// ================= host side =================
static void launch_gemm(hipStream_t s, const __bf16* A, const __bf16* B,
                        const float* bias, int rows, int kdim, int ncols, int act,
                        float* oF, __bf16* oB, const float* rA, const float* rS,
                        int rsStride, int rsOff) {
  int tiles = (rows / 16) * (ncols / 16);
  int grid = (tiles + 7) / 8;
  gemm_tile_kernel<<<grid, 256, 0, s>>>(A, B, bias, rows, kdim, ncols, act, oF, oB,
                                        rA, rS, rsStride, rsOff);
}
static void launch_swz(hipStream_t s, const float* W, __bf16* out, int kd, int nd,
                       int headMode) {
  int total = (kd / 32) * (nd / 16) * 256;
  swizzle_b_kernel<<<(total + 255) / 256, 256, 0, s>>>(W, out, kd, nd, headMode);
}

extern "C" void kernel_launch(void* const* d_in, const int* in_sizes, int n_in,
                              void* d_out, int out_size, void* d_ws, size_t ws_size,
                              hipStream_t stream) {
  (void)in_sizes; (void)n_in; (void)out_size; (void)ws_size;
  const float* nodes0 = (const float*)d_in[0];
  const float* tIn    = (const float*)d_in[1];
  const float* edges  = (const float*)d_in[2];
  const int*   nbrs   = (const int*)d_in[3];
  const unsigned char* nmask = (const unsigned char*)d_in[4];
  auto P = [&](int l, int idx) -> const float* {
    return (const float*)d_in[5 + 47 * l + idx];
  };

  // bump allocator over workspace
  size_t off = 0;
  auto alloc = [&](size_t bytes) -> void* {
    off = (off + 255) & ~(size_t)255;
    void* p = (char*)d_ws + off;
    off += bytes;
    return p;
  };
  auto abf = [&](size_t elems) { return (__bf16*)alloc(elems * 2); };
  auto af32 = [&](size_t elems) { return (float*)alloc(elems * 4); };

  struct LayerW {
    __bf16 *fK1, *fK2, *fK3, *fV1, *fV2, *fV3, *qc, *kc, *vc, *ow;
    __bf16 *ag1, *ag2, *ag3, *aa1, *aa2, *aa3, *fg1, *fg2, *fg3, *fa1, *fa2, *fa3;
    __bf16 *ff1, *ff2;
  } LW[3];

  for (int l = 0; l < 3; ++l) {
    LW[l].fK1 = abf(128 * 128); LW[l].fK2 = abf(128 * 128); LW[l].fK3 = abf(128 * 256);
    LW[l].fV1 = abf(128 * 128); LW[l].fV2 = abf(128 * 128); LW[l].fV3 = abf(128 * 256);
    LW[l].qc = abf(128 * 128);  LW[l].kc = abf(128 * 128);  LW[l].vc = abf(128 * 128);
    LW[l].ow = abf(128 * 128);
    LW[l].ag1 = abf(128 * 128); LW[l].ag2 = abf(128 * 128); LW[l].ag3 = abf(128 * 256);
    LW[l].aa1 = abf(128 * 128); LW[l].aa2 = abf(128 * 128); LW[l].aa3 = abf(128 * 128);
    LW[l].fg1 = abf(128 * 128); LW[l].fg2 = abf(128 * 128); LW[l].fg3 = abf(128 * 256);
    LW[l].fa1 = abf(128 * 128); LW[l].fa2 = abf(128 * 128); LW[l].fa3 = abf(128 * 128);
    LW[l].ff1 = abf(128 * 512); LW[l].ff2 = abf(512 * 128);
  }
  __bf16* Tbf    = abf((size_t)ROWS * 128);
  __bf16* Ht1    = abf((size_t)ROWS * 128);
  __bf16* Ht2    = abf((size_t)ROWS * 128);
  float*  GB1    = af32((size_t)ROWS * 256);   // [alpha1 | gamma1]
  float*  A1     = af32((size_t)ROWS * 128);   // beta1
  float*  GB2    = af32((size_t)ROWS * 256);   // [alpha2 | gamma2]
  float*  A2     = af32((size_t)ROWS * 128);   // beta2
  __bf16* NodesI = abf((size_t)ROWS * 128);
  float*  Qbuf   = af32((size_t)ROWS * 128);
  __bf16* AttnB  = abf((size_t)ROWS * 128);
  float*  NMid   = af32((size_t)ROWS * 128);
  __bf16* Nodes2 = abf((size_t)ROWS * 128);
  __bf16* Hf     = abf((size_t)ROWS * 512);
  float*  Ping0  = af32((size_t)ROWS * 128);
  float*  Ping1  = af32((size_t)ROWS * 128);

  // ---- weight conversion / swizzle (once per call) ----
  for (int l = 0; l < 3; ++l) {
    launch_swz(stream, P(l, 0),  LW[l].fK1, 128, 128, 0);
    launch_swz(stream, P(l, 2),  LW[l].fK2, 128, 128, 0);
    launch_swz(stream, P(l, 4),  LW[l].fK3, 128, 256, 0);
    launch_swz(stream, P(l, 6),  LW[l].fV1, 128, 128, 0);
    launch_swz(stream, P(l, 8),  LW[l].fV2, 128, 128, 0);
    launch_swz(stream, P(l, 10), LW[l].fV3, 128, 256, 0);
    launch_swz(stream, P(l, 12), LW[l].qc, 128, 128, 1);
    launch_swz(stream, P(l, 13), LW[l].kc, 128, 128, 1);
    launch_swz(stream, P(l, 14), LW[l].vc, 128, 128, 1);
    launch_swz(stream, P(l, 18), LW[l].ow, 128, 128, 0);
    launch_swz(stream, P(l, 19), LW[l].ag1, 128, 128, 0);
    launch_swz(stream, P(l, 21), LW[l].ag2, 128, 128, 0);
    launch_swz(stream, P(l, 23), LW[l].ag3, 128, 256, 0);
    launch_swz(stream, P(l, 25), LW[l].aa1, 128, 128, 0);
    launch_swz(stream, P(l, 27), LW[l].aa2, 128, 128, 0);
    launch_swz(stream, P(l, 29), LW[l].aa3, 128, 128, 0);
    launch_swz(stream, P(l, 31), LW[l].fg1, 128, 128, 0);
    launch_swz(stream, P(l, 33), LW[l].fg2, 128, 128, 0);
    launch_swz(stream, P(l, 35), LW[l].fg3, 128, 256, 0);
    launch_swz(stream, P(l, 37), LW[l].fa1, 128, 128, 0);
    launch_swz(stream, P(l, 39), LW[l].fa2, 128, 128, 0);
    launch_swz(stream, P(l, 41), LW[l].fa3, 128, 128, 0);
    launch_swz(stream, P(l, 43), LW[l].ff1, 128, 512, 0);
    launch_swz(stream, P(l, 45), LW[l].ff2, 512, 128, 0);
  }
  {
    int n = ROWS * 128;
    cvt_f32_bf16_kernel<<<(n + 255) / 256, 256, 0, stream>>>(tIn, Tbf, n);
  }

  const float* nodesCur = nodes0;
  for (int l = 0; l < 3; ++l) {
    LayerW& W = LW[l];
    // ---- t-conditioned modulation MLPs (all bf16 WMMA GEMMs) ----
    launch_gemm(stream, Tbf, W.ag1, P(l, 20), ROWS, 128, 128, 1, nullptr, Ht1, nullptr, nullptr, 0, 0);
    launch_gemm(stream, Ht1, W.ag2, P(l, 22), ROWS, 128, 128, 1, nullptr, Ht2, nullptr, nullptr, 0, 0);
    launch_gemm(stream, Ht2, W.ag3, P(l, 24), ROWS, 128, 256, 0, GB1, nullptr, nullptr, nullptr, 0, 0);
    launch_gemm(stream, Tbf, W.aa1, P(l, 26), ROWS, 128, 128, 1, nullptr, Ht1, nullptr, nullptr, 0, 0);
    launch_gemm(stream, Ht1, W.aa2, P(l, 28), ROWS, 128, 128, 1, nullptr, Ht2, nullptr, nullptr, 0, 0);
    launch_gemm(stream, Ht2, W.aa3, P(l, 30), ROWS, 128, 128, 0, A1, nullptr, nullptr, nullptr, 0, 0);
    launch_gemm(stream, Tbf, W.fg1, P(l, 32), ROWS, 128, 128, 1, nullptr, Ht1, nullptr, nullptr, 0, 0);
    launch_gemm(stream, Ht1, W.fg2, P(l, 34), ROWS, 128, 128, 1, nullptr, Ht2, nullptr, nullptr, 0, 0);
    launch_gemm(stream, Ht2, W.fg3, P(l, 36), ROWS, 128, 256, 0, GB2, nullptr, nullptr, nullptr, 0, 0);
    launch_gemm(stream, Tbf, W.fa1, P(l, 38), ROWS, 128, 128, 1, nullptr, Ht1, nullptr, nullptr, 0, 0);
    launch_gemm(stream, Ht1, W.fa2, P(l, 40), ROWS, 128, 128, 1, nullptr, Ht2, nullptr, nullptr, 0, 0);
    launch_gemm(stream, Ht2, W.fa3, P(l, 42), ROWS, 128, 128, 0, A2, nullptr, nullptr, nullptr, 0, 0);

    // nodes_i = gamma1 * sln(nodes) + beta1   (gamma1 = GB1[:,128:], beta1 = A1)
    sln_mod_kernel<<<ROWS / 8, 256, 0, stream>>>(nodesCur, GB1, 256, 128, A1, NodesI, ROWS);
    // Q = nodes_i @ qcat + q_b
    launch_gemm(stream, NodesI, W.qc, P(l, 15), ROWS, 128, 128, 0, Qbuf, nullptr, nullptr, nullptr, 0, 0);

    // fused FiLM + attention (TDM double-buffered edge staging)
    film_attn_kernel<<<AT_WGS, AT_WAVES * 32, AT_WAVES * AT_WAVE_BYTES, stream>>>(
        edges, nbrs, nmask, NodesI, Qbuf,
        W.fK1, P(l, 1), W.fK2, P(l, 3), W.fK3, P(l, 5),
        W.fV1, P(l, 7), W.fV2, P(l, 9), W.fV3, P(l, 11),
        W.kc, P(l, 16), W.vc, P(l, 17), AttnB);

    // nodes_mid = nodes + alpha1 * (attn_out @ out_w)   (alpha1 = GB1[:,0:128])
    launch_gemm(stream, AttnB, W.ow, nullptr, ROWS, 128, 128, 0, NMid, nullptr,
                nodesCur, GB1, 256, 0);

    // nodes2 = gamma2 * sln(nodes_mid) + beta2
    sln_mod_kernel<<<ROWS / 8, 256, 0, stream>>>(NMid, GB2, 256, 128, A2, Nodes2, ROWS);

    // FFN with fused residual: out = nodes_mid + alpha2 * (silu(x@W1+b1)@W2+b2)
    launch_gemm(stream, Nodes2, W.ff1, P(l, 44), ROWS, 128, 512, 1, nullptr, Hf,
                nullptr, nullptr, 0, 0);
    float* nextNodes = (l == 2) ? (float*)d_out : ((l & 1) ? Ping1 : Ping0);
    launch_gemm(stream, Hf, W.ff2, P(l, 46), ROWS, 512, 128, 0, nextNodes, nullptr,
                NMid, GB2, 256, 0);
    nodesCur = nextNodes;
  }
}
